// semidual_23965917512075
// MI455X (gfx1250) — matrix operand order repaired and checked by reading, not compiled, not run on hardware
//
#include <hip/hip_runtime.h>

// Problem constants (match the reference).
#define N_ROWS   32768
#define M_COLS   8192
#define DIM      64
#define BLK_ROWS 128                 // rows per block = 8 waves * 16
#define JC       64                  // Y rows staged per LDS chunk
#define NCHUNK   (M_COLS / JC)       // 128
#define YSTR     72                  // padded LDS row stride (ushorts); 144B, 16B-aligned segs

typedef __attribute__((ext_vector_type(16))) __bf16        v16bf;
typedef __attribute__((ext_vector_type(16))) unsigned short u16x16;
typedef __attribute__((ext_vector_type(8)))  unsigned short u16x8;
typedef __attribute__((ext_vector_type(8)))  float          v8f;

// ---- bf16 split helpers (hi = truncation -> exact upper bits; lo = RNE of residual) ----
__device__ __forceinline__ unsigned short bf16_hi_bits(float f) {
  union { float f; unsigned u; } v; v.f = f;
  return (unsigned short)(v.u >> 16);
}
__device__ __forceinline__ float bf16_bits_to_f32(unsigned short b) {
  union { float f; unsigned u; } v; v.u = ((unsigned)b) << 16;
  return v.f;
}
__device__ __forceinline__ unsigned short bf16_rne_bits(float f) {
  union { float f; unsigned u; } v; v.f = f;
  unsigned u = v.u;
  return (unsigned short)((u + 0x7FFFu + ((u >> 16) & 1u)) >> 16);
}

// ---- gfx1250 async global->LDS copy (ASYNCcnt-tracked, no VGPR round trip) ----
__device__ __forceinline__ void async_copy_b128(unsigned lds_off, const void* gaddr) {
  asm volatile("global_load_async_to_lds_b128 %0, %1, off"
               :: "v"(lds_off), "v"(gaddr) : "memory");
}
__device__ __forceinline__ void async_copy_b32(unsigned lds_off, const void* gaddr) {
  asm volatile("global_load_async_to_lds_b32 %0, %1, off"
               :: "v"(lds_off), "v"(gaddr) : "memory");
}
__device__ __forceinline__ void wait_async_zero() {
  asm volatile("s_wait_asynccnt 0" ::: "memory");
}

// ---------------------------------------------------------------------------
// Kernel 0: one-time prep.
//   T[j]  = ||y_j||^2 - psi[j]
//   YH/YL = bf16 hi/lo split planes of Y (row-major, 64 ushorts per row).
// Doing the split once here removes ~100 conversion VALU ops per tile from the
// 256 blocks of the hot kernel (they were re-splitting Y redundantly).
// ---------------------------------------------------------------------------
__global__ __launch_bounds__(256) void prep_kernel(
    const float* __restrict__ Y, const float* __restrict__ psi,
    float* __restrict__ T, unsigned short* __restrict__ YH,
    unsigned short* __restrict__ YL)
{
  int j = blockIdx.x * 256 + threadIdx.x;
  const float4* yr = (const float4*)(Y + (size_t)j * DIM);
  float s = 0.f;
#pragma unroll
  for (int i = 0; i < DIM / 4; ++i) {
    float4 v = yr[i];
    float fv[4] = { v.x, v.y, v.z, v.w };
    unsigned short hb[4], lb[4];
#pragma unroll
    for (int e = 0; e < 4; ++e) {
      s += fv[e] * fv[e];
      hb[e] = bf16_hi_bits(fv[e]);
      lb[e] = bf16_rne_bits(fv[e] - bf16_bits_to_f32(hb[e]));
    }
    uint2 hp = { (unsigned)hb[0] | ((unsigned)hb[1] << 16),
                 (unsigned)hb[2] | ((unsigned)hb[3] << 16) };
    uint2 lp = { (unsigned)lb[0] | ((unsigned)lb[1] << 16),
                 (unsigned)lb[2] | ((unsigned)lb[3] << 16) };
    *(uint2*)(YH + (size_t)j * DIM + i * 4) = hp;
    *(uint2*)(YL + (size_t)j * DIM + i * 4) = lp;
  }
  T[j] = s - psi[j];
}

// ---------------------------------------------------------------------------
// Kernel 1: fused  min_j ( t[j] - 2 * <x_i, y_j> )  per row, bf16x3 WMMA,
// double-buffered async global->LDS staging of the pre-split Y planes.
// ---------------------------------------------------------------------------
__global__ __launch_bounds__(256) void semidual_wmma_kernel(
    const float* __restrict__ X,
    const unsigned short* __restrict__ YH, const unsigned short* __restrict__ YL,
    const float* __restrict__ T, float* __restrict__ partials)
{
  __shared__ unsigned short YsH[2][JC][YSTR];
  __shared__ unsigned short YsL[2][JC][YSTR];
  __shared__ float Ts[2][JC];
  __shared__ float wsum[8];

  const int tid  = threadIdx.x;
  const int wave = tid >> 5;
  const int lane = tid & 31;
  const int hi   = lane >> 4;    // half-wave id (K-interleave group)
  const int nres = lane & 15;    // N residue within a tile

  // LDS byte offsets of the staging buffers (ISA 10.2).
  const unsigned ldsH[2] = { (unsigned)(size_t)&YsH[0][0][0],
                             (unsigned)(size_t)&YsH[1][0][0] };
  const unsigned ldsL[2] = { (unsigned)(size_t)&YsL[0][0][0],
                             (unsigned)(size_t)&YsL[1][0][0] };
  const unsigned ldsT[2] = { (unsigned)(size_t)&Ts[0][0],
                             (unsigned)(size_t)&Ts[1][0] };

  // ---- A fragments: 16 rows x 64 cols per wave, bf16 hi/lo, ISA 16-bit A layout.
  // lane (hi,nres): M = nres; vector elem e<8 -> K = c*32 + 8*hi + e
  //                            elem e>=8 -> K = c*32 + 8*hi + e + 8
  const int row = blockIdx.x * BLK_ROWS + wave * 16 + nres;
  const float* xrow = X + (size_t)row * DIM;

  v16bf a_hi[2], a_lo[2];
  float x2part = 0.f;
#pragma unroll
  for (int c = 0; c < 2; ++c) {
    u16x16 h, l;
#pragma unroll
    for (int e = 0; e < 16; ++e) {
      int k = c * 32 + 8 * hi + ((e < 8) ? e : (e + 8));
      float f = xrow[k];
      x2part += f * f;
      unsigned short hb = bf16_hi_bits(f);
      h[e] = hb;
      l[e] = bf16_rne_bits(f - bf16_bits_to_f32(hb));
    }
    a_hi[c] = __builtin_bit_cast(v16bf, h);
    a_lo[c] = __builtin_bit_cast(v16bf, l);
  }
  // Each lane summed half of its row's squares; other half lives across the xor-16 lane.
  float x2full = x2part + __shfl_xor(x2part, 16, 32);

  float q[8];
#pragma unroll
  for (int r = 0; r < 8; ++r) q[r] = 3.402823466e38f;

  // Async-stage one chunk: 2 planes * 64 rows * 8 x 16B segments = 2 per thread
  // per plane, plus the t[] tile. Issued on the ASYNCcnt path.
  auto stage_chunk = [&](int chunk, int buf) {
#pragma unroll
    for (int p = 0; p < 2; ++p) {
      int i  = p * 256 + tid;      // 0..511
      int r  = i >> 3;             // chunk row
      int sg = i & 7;              // 16B segment within the 128B row
      unsigned dofs = (unsigned)(r * (YSTR * 2) + sg * 16);
      const void* gH = YH + (size_t)(chunk * JC + r) * DIM + sg * 8;
      const void* gL = YL + (size_t)(chunk * JC + r) * DIM + sg * 8;
      async_copy_b128(ldsH[buf] + dofs, gH);
      async_copy_b128(ldsL[buf] + dofs, gL);
    }
    if (tid < JC)
      async_copy_b32(ldsT[buf] + tid * 4, T + chunk * JC + tid);
  };

  stage_chunk(0, 0);   // prologue: chunk 0 in flight

  for (int chunk = 0; chunk < NCHUNK; ++chunk) {
    const int buf = chunk & 1;
    wait_async_zero();     // this wave's staged data (chunk) has landed in LDS
    __syncthreads();       // all waves' shares landed; prev compute done

    if (chunk + 1 < NCHUNK) stage_chunk(chunk + 1, buf ^ 1);

#pragma unroll
    for (int js = 0; js < JC / 16; ++js) {
      const unsigned short* yh = &YsH[buf][js * 16 + nres][0];
      const unsigned short* yl = &YsL[buf][js * 16 + nres][0];
      float tj = Ts[buf][js * 16 + nres];

      v8f acc = {};
#pragma unroll
      for (int c = 0; c < 2; ++c) {
        // B fragment (ISA 16-bit B layout): lane (hi,nres): N = nres,
        // elem e<8 -> K = c*32 + 8*hi + e, elem e>=8 -> K = c*32 + 8*hi + e + 8
        int off = c * 32 + 8 * hi;
        u16x8 h0 = *(const u16x8*)&yh[off];
        u16x8 h1 = *(const u16x8*)&yh[off + 16];
        u16x8 l0 = *(const u16x8*)&yl[off];
        u16x8 l1 = *(const u16x8*)&yl[off + 16];
        u16x16 bhu = __builtin_shufflevector(h0, h1, 0,1,2,3,4,5,6,7,8,9,10,11,12,13,14,15);
        u16x16 blu = __builtin_shufflevector(l0, l1, 0,1,2,3,4,5,6,7,8,9,10,11,12,13,14,15);
        v16bf vbh = __builtin_bit_cast(v16bf, bhu);
        v16bf vbl = __builtin_bit_cast(v16bf, blu);
        // bf16x3: hi*hi + hi*lo + lo*hi  (lo*lo below fp32 ulp)
        acc = __builtin_amdgcn_wmma_f32_16x16x32_bf16(false, a_hi[c], false, vbh,
                                                      (short)0, acc, false, false);
        acc = __builtin_amdgcn_wmma_f32_16x16x32_bf16(false, a_hi[c], false, vbl,
                                                      (short)0, acc, false, false);
        acc = __builtin_amdgcn_wmma_f32_16x16x32_bf16(false, a_lo[c], false, vbh,
                                                      (short)0, acc, false, false);
      }
      // cost(i,j) - x2_i = t[j] - 2*dot ; x2 added after the min (constant per row).
#pragma unroll
      for (int r = 0; r < 8; ++r)
        q[r] = fminf(q[r], __builtin_fmaf(-2.0f, acc[r], tj));
    }
    // next iteration's wait+barrier protects both buffers; no trailing barrier
  }

  // ---- reduce min across the 16 N-residue lanes of each half-wave.
#pragma unroll
  for (int s = 1; s < 16; s <<= 1) {
#pragma unroll
    for (int r = 0; r < 8; ++r)
      q[r] = fminf(q[r], __shfl_xor(q[r], s, 32));
  }

  // ---- per-wave sum of the 16 row minima (+ x2 per row), then block partial.
  float s = 0.f;
#pragma unroll
  for (int r = 0; r < 8; ++r) {
    float x2r = __shfl(x2full, r + 8 * hi, 32);  // x2 of row (r+8*hi) lives at lane nres==r+8*hi
    s += q[r] + x2r;
  }
  s += __shfl_xor(s, 16, 32);    // combine rows 0-7 half with rows 8-15 half

  if (lane == 0) wsum[wave] = s;
  __syncthreads();
  if (tid == 0) {
    float b = 0.f;
#pragma unroll
    for (int w = 0; w < 8; ++w) b += wsum[w];
    partials[blockIdx.x] = b;
  }
}

// ---------------------------------------------------------------------------
// Kernel 2: loss = (1/N) * sum(partials) + (1/M) * sum(psi)
// ---------------------------------------------------------------------------
__global__ __launch_bounds__(256) void finalize_kernel(
    const float* __restrict__ partials, const float* __restrict__ psi,
    float* __restrict__ out)
{
  __shared__ float sh[256];
  int tid = threadIdx.x;
  float a = partials[tid];                      // exactly 256 partials
  float b = 0.f;
  for (int j = tid; j < M_COLS; j += 256) b += psi[j];
  sh[tid] = a * (1.0f / N_ROWS) + b * (1.0f / M_COLS);
  __syncthreads();
  for (int s = 128; s > 0; s >>= 1) {
    if (tid < s) sh[tid] += sh[tid + s];
    __syncthreads();
  }
  if (tid == 0) out[0] = sh[0];
}

// ---------------------------------------------------------------------------
extern "C" void kernel_launch(void* const* d_in, const int* in_sizes, int n_in,
                              void* d_out, int out_size, void* d_ws, size_t ws_size,
                              hipStream_t stream) {
  const float* X   = (const float*)d_in[0];   // [N, D]
  const float* Y   = (const float*)d_in[1];   // [M, D]
  const float* psi = (const float*)d_in[2];   // [M]

  char* ws = (char*)d_ws;
  float*          T        = (float*)(ws);                       // [M]
  float*          partials = (float*)(ws + 32768);               // [256]
  unsigned short* YH       = (unsigned short*)(ws + 65536);      // [M*D] bf16 hi bits
  unsigned short* YL       = (unsigned short*)(ws + 65536 + (size_t)M_COLS * DIM * 2); // lo bits

  prep_kernel<<<M_COLS / 256, 256, 0, stream>>>(Y, psi, T, YH, YL);
  semidual_wmma_kernel<<<N_ROWS / BLK_ROWS, 256, 0, stream>>>(X, YH, YL, T, partials);
  finalize_kernel<<<1, 256, 0, stream>>>(partials, psi, (float*)d_out);
}
